// MultiHeadAttention_36017595744736
// MI455X (gfx1250) — matrix operand order repaired and checked by reading, not compiled, not run on hardware
//
#include <hip/hip_runtime.h>

typedef __bf16 bf16_t;
typedef __attribute__((ext_vector_type(16))) __bf16 v16bf;
typedef __attribute__((ext_vector_type(8)))  __bf16 v8bf;
typedef __attribute__((ext_vector_type(8)))  float  v8f;

#define NB   32
#define NT   512
#define NC   256
#define NH   8
#define ND   32

// ---------------- WMMA fragment loaders (CDNA5 16x16x32 bf16 layouts) ----------------
// A (16xK=32, M x K): lane&15 = M row; lane>>4 selects K base {0,8};
//   elems 0..7  -> K = base..base+7      (16B chunk)
//   elems 8..15 -> K = 16+base..16+base+7 (16B chunk at +16 elems)
static __device__ __forceinline__ v16bf load_a_frag(const bf16_t* __restrict__ base,
                                                    int stride, int m0, int k0, int lane) {
  const bf16_t* p = base + (size_t)(m0 + (lane & 15)) * stride + k0 + ((lane >> 4) << 3);
  union { v16bf v; v8bf h[2]; } u;
  u.h[0] = *(const v8bf*)(p);
  u.h[1] = *(const v8bf*)(p + 16);
  return u.v;
}
// B (K=32 x 16, stored as [n][k] contiguous-in-k): lane&15 = N col; lane>>4 selects K base {0,8};
//   elems 0..7 -> K = base..base+7, elems 8..15 -> K = 16+base..16+base+7
static __device__ __forceinline__ v16bf load_b_frag(const bf16_t* __restrict__ base,
                                                    int stride, int n0, int k0, int lane) {
  const bf16_t* p = base + (size_t)(n0 + (lane & 15)) * stride + k0 + ((lane >> 4) << 3);
  union { v16bf v; v8bf h[2]; } u;
  u.h[0] = *(const v8bf*)(p);
  u.h[1] = *(const v8bf*)(p + 16);
  return u.v;
}

// cross-lane reductions over the 16 lanes that hold one C/D row (xor masks stay in half-wave)
static __device__ __forceinline__ float rmax16(float x) {
  #pragma unroll
  for (int m = 1; m < 16; m <<= 1) x = fmaxf(x, __shfl_xor(x, m, 32));
  return x;
}
static __device__ __forceinline__ float rsum16(float x) {
  #pragma unroll
  for (int m = 1; m < 16; m <<= 1) x += __shfl_xor(x, m, 32);
  return x;
}

// ---------------- gfx1250 async global->LDS (ASYNCcnt-tracked DMA), ISA 08_async_tensor ----------------
// GV mode: per-lane 64-bit global address, per-lane 32-bit LDS byte offset (wave-relative).
static __device__ __forceinline__ void async_ld_b128(unsigned lds_off, const bf16_t* gaddr) {
  asm volatile("global_load_async_to_lds_b128 %0, %1, off"
               :: "v"(lds_off), "v"((unsigned long long)(const void*)gaddr)
               : "memory");
}
static __device__ __forceinline__ void wait_async_le2() {
  asm volatile("s_wait_asynccnt 0x2" ::: "memory");
}
static __device__ __forceinline__ void wait_async_0() {
  asm volatile("s_wait_asynccnt 0x0" ::: "memory");
}
// generic LDS pointer -> wave-relative LDS byte offset (flat LDS addr keeps offset in [31:0])
static __device__ __forceinline__ unsigned lds_off_of(const void* p) {
  return (unsigned)(size_t)p;
}

// ---------------- conversion / transpose pre-pass ----------------
__global__ void cvt_f32_bf16(const float* __restrict__ src, bf16_t* __restrict__ dst, int n) {
  int i = blockIdx.x * blockDim.x + threadIdx.x;
  if (i < n) dst[i] = (bf16_t)src[i];
}
// Wt[n][k] = W[k][n], 256x256
__global__ void wt_transpose(const float* __restrict__ W, bf16_t* __restrict__ Wt) {
  int idx = blockIdx.x * blockDim.x + threadIdx.x;
  int k = idx & 255;
  int n = idx >> 8;
  Wt[n * NC + k] = (bf16_t)W[k * NC + n];
}

// ---------------- fused QKV projection: x[16384,256] @ W -> Q/K [bh][t][d], V^T [bh][d][t] ----------------
__global__ __launch_bounds__(32) void qkv_gemm(const bf16_t* __restrict__ xb,
                                               const bf16_t* __restrict__ Wqt,
                                               const bf16_t* __restrict__ Wkt,
                                               const bf16_t* __restrict__ Wvt,
                                               bf16_t* __restrict__ Qb,
                                               bf16_t* __restrict__ Kb,
                                               bf16_t* __restrict__ Vt) {
  const int mt    = blockIdx.x;        // 0..1023 : 16-row tile of x
  const int strip = blockIdx.y;        // 0..11   : weight (y/4) x 64-col strip (y%4)
  const int w     = strip >> 2;
  const int n0    = (strip & 3) * 64;
  const int lane  = threadIdx.x;
  const bf16_t* Wt = (w == 0) ? Wqt : (w == 1) ? Wkt : Wvt;

  v8f acc[4] = {{}, {}, {}, {}};
  for (int k0 = 0; k0 < NC; k0 += 32) {
    v16bf a = load_a_frag(xb, NC, mt * 16, k0, lane);
    #pragma unroll
    for (int j = 0; j < 4; ++j) {
      v16bf b = load_b_frag(Wt, NC, n0 + j * 16, k0, lane);
      acc[j] = __builtin_amdgcn_wmma_f32_16x16x32_bf16(false, a, false, b, (short)0, acc[j],
                                                       false, false);
    }
  }
  const int mb = (lane >> 4) << 3;     // C/D: row = v + mb, col = lane&15
  const int n  = lane & 15;
  #pragma unroll
  for (int j = 0; j < 4; ++j) {
    #pragma unroll
    for (int v = 0; v < 8; ++v) {
      int row = mt * 16 + mb + v;      // global x row = b*T + t
      int col = n0 + j * 16 + n;       // h*D + d
      int b = row >> 9, t = row & 511;
      int h = col >> 5, d = col & 31;
      size_t bh = (size_t)b * NH + h;
      bf16_t val = (bf16_t)acc[j][v];
      if (w == 2) Vt[(bh * ND + d) * NT + t] = val;           // V transposed
      else        ((w == 0) ? Qb : Kb)[(bh * NT + t) * ND + d] = val;
    }
  }
}

// ---------------- attention ----------------
// 4 waves / block; block covers 64 queries of one (b,h). K/V pair-tiles (32 keys) are
// DMA'd into double-buffered LDS with global_load_async_to_lds_b128 and shared by all
// 4 waves; each wave runs online softmax for its own 16-query tile.
#define KVPAD 40           // LDS row pitch (bf16) for 32-wide tiles: 80B, conflict-free b128 reads

__global__ __launch_bounds__(128) void attn_kernel(const bf16_t* __restrict__ Qb,
                                                   const bf16_t* __restrict__ Kb,
                                                   const bf16_t* __restrict__ Vt,
                                                   bf16_t* __restrict__ att) {
  const int bh  = blockIdx.x;          // 0..255 = b*H + h
  const int qb  = blockIdx.y;          // 0..7   : 64-query block
  const int tid = threadIdx.x;         // 0..127
  const int lane = tid & 31;
  const int wid  = tid >> 5;           // wave id 0..3
  const int qt   = qb * 4 + wid;       // this wave's 16-query tile
  const int h    = bh & (NH - 1);
  const float slope = exp2f(-(float)(h + 1));   // ALiBi slopes for H=8: 2^-(h+1)
  const float scale = 0.17677669529663687f;     // 1/sqrt(32)

  const bf16_t* Qh = Qb + (size_t)bh * NT * ND;
  const bf16_t* Kh = Kb + (size_t)bh * NT * ND;
  const bf16_t* Vh = Vt + (size_t)bh * ND * NT;

  __shared__ __attribute__((aligned(16))) bf16_t Kbuf[2][32 * KVPAD]; // [key 0..31][d 0..31]
  __shared__ __attribute__((aligned(16))) bf16_t Vbuf[2][32 * KVPAD]; // [d 0..31][key 0..31]
  __shared__ __attribute__((aligned(16))) bf16_t Plds[4][16 * KVPAD]; // per-wave P tile 16x32

  // per-thread DMA slot: 128 threads x 16B = one 32x32 bf16 tile per issue pair
  const int drow = tid >> 2;           // 0..31
  const int dcol = tid & 3;            // 16B chunk within 64B row
  const unsigned lK0 = lds_off_of(&Kbuf[0][0]) + (unsigned)(drow * (KVPAD * 2) + dcol * 16);
  const unsigned lK1 = lds_off_of(&Kbuf[1][0]) + (unsigned)(drow * (KVPAD * 2) + dcol * 16);
  const unsigned lV0 = lds_off_of(&Vbuf[0][0]) + (unsigned)(drow * (KVPAD * 2) + dcol * 16);
  const unsigned lV1 = lds_off_of(&Vbuf[1][0]) + (unsigned)(drow * (KVPAD * 2) + dcol * 16);

  const v16bf aq = load_a_frag(Qh, ND, qt * 16, 0, lane);   // Q fragment (K-dim = D = 32)

  const int mb = (lane >> 4) << 3;
  const int n  = lane & 15;

  float run_max[8], run_sum[8];
  #pragma unroll
  for (int v = 0; v < 8; ++v) { run_max[v] = -1e30f; run_sum[v] = 0.f; }
  v8f o0 = {}, o1 = {};

  const int ktend = qb * 4 + 3;        // last key tile any wave in this block needs (odd)

  // prologue: stage pair-tile kt=0 into buffer 0 (2 async ops / wave)
  async_ld_b128(lK0, Kh + (size_t)(0 * 16 + drow) * ND + dcol * 8);
  async_ld_b128(lV0, Vh + (size_t)drow * NT + 0 * 16 + dcol * 8);

  int cur = 0;
  for (int kt = 0; kt <= ktend; kt += 2) {
    const bool pf = (kt + 2) <= ktend;
    if (pf) {                          // prefetch next pair-tile into the other buffer
      async_ld_b128(cur ? lK0 : lK1, Kh + (size_t)((kt + 2) * 16 + drow) * ND + dcol * 8);
      async_ld_b128(cur ? lV0 : lV1, Vh + (size_t)drow * NT + (kt + 2) * 16 + dcol * 8);
      wait_async_le2();                // drain current stage, keep prefetch in flight
    } else {
      wait_async_0();
    }
    __syncthreads();                   // current buffer visible to all 4 waves

    const bf16_t* Kt = &Kbuf[cur][0];
    const bf16_t* Vb = &Vbuf[cur][0];

    // ---- S = Q K^T : two 16x16 score tiles from the shared LDS K tile ----
    v16bf bk0 = load_b_frag(Kt, KVPAD, 0, 0, lane);
    v16bf bk1 = load_b_frag(Kt, KVPAD, 16, 0, lane);
    v8f c0 = {}, c1 = {};
    c0 = __builtin_amdgcn_wmma_f32_16x16x32_bf16(false, aq, false, bk0, (short)0, c0, false, false);
    c1 = __builtin_amdgcn_wmma_f32_16x16x32_bf16(false, aq, false, bk1, (short)0, c1, false, false);

    // ---- scale + ALiBi bias + causal mask + online softmax ----
    bf16_t* Pw = &Plds[wid][0];
    #pragma unroll
    for (int v = 0; v < 8; ++v) {
      int rt = qt * 16 + mb + v;                     // absolute query position
      int s0 = kt * 16 + n;                          // absolute key positions
      int s1 = s0 + 16;
      float x0 = (s0 <= rt) ? c0[v] * scale + slope * (float)(s0 - rt) : -1e30f;
      float x1 = (s1 <= rt) ? c1[v] * scale + slope * (float)(s1 - rt) : -1e30f;
      float nm = fmaxf(run_max[v], rmax16(fmaxf(x0, x1)));
      float corr = __expf(run_max[v] - nm);
      run_max[v] = nm;
      float p0 = __expf(x0 - nm);
      float p1 = __expf(x1 - nm);
      run_sum[v] = run_sum[v] * corr + rsum16(p0 + p1);
      o0[v] *= corr;
      o1[v] *= corr;
      Pw[(mb + v) * KVPAD + n]      = (bf16_t)p0;    // stage P (A-matrix of PV)
      Pw[(mb + v) * KVPAD + 16 + n] = (bf16_t)p1;
    }

    // ---- O += P V : P from this wave's LDS region, V from the shared LDS V^T tile ----
    {
      const bf16_t* pp = Pw + (size_t)(lane & 15) * KVPAD + ((lane >> 4) << 3);
      union { v16bf v; v8bf hh[2]; } u;
      u.hh[0] = *(const v8bf*)(pp);
      u.hh[1] = *(const v8bf*)(pp + 16);
      v16bf bv0 = load_b_frag(Vb, KVPAD, 0, 0, lane);    // d = 0..15
      v16bf bv1 = load_b_frag(Vb, KVPAD, 16, 0, lane);   // d = 16..31
      o0 = __builtin_amdgcn_wmma_f32_16x16x32_bf16(false, u.v, false, bv0, (short)0, o0, false, false);
      o1 = __builtin_amdgcn_wmma_f32_16x16x32_bf16(false, u.v, false, bv1, (short)0, o1, false, false);
    }
    __syncthreads();                   // everyone done reading buf[cur] before it is re-filled
    cur ^= 1;
  }

  // ---- normalize and scatter to att[(b*T+t)][h*D+d] (bf16 for the output GEMM) ----
  #pragma unroll
  for (int v = 0; v < 8; ++v) {
    float inv = 1.0f / run_sum[v];
    int t = qt * 16 + mb + v;
    size_t rowbase = ((size_t)((bh >> 3) * NT + t)) * NC + (size_t)h * ND;
    att[rowbase + n]      = (bf16_t)(o0[v] * inv);
    att[rowbase + 16 + n] = (bf16_t)(o1[v] * inv);
  }
}

// ---------------- output projection: att[16384,256] @ Wo + bo -> f32 ----------------
__global__ __launch_bounds__(32) void out_gemm(const bf16_t* __restrict__ att,
                                               const bf16_t* __restrict__ Wot,
                                               const float* __restrict__ bo,
                                               float* __restrict__ out) {
  const int mt   = blockIdx.x;        // 0..1023
  const int n0   = blockIdx.y * 64;   // 0..3 strips
  const int lane = threadIdx.x;
  v8f acc[4] = {{}, {}, {}, {}};
  for (int k0 = 0; k0 < NC; k0 += 32) {
    v16bf a = load_a_frag(att, NC, mt * 16, k0, lane);
    #pragma unroll
    for (int j = 0; j < 4; ++j) {
      v16bf b = load_b_frag(Wot, NC, n0 + j * 16, k0, lane);
      acc[j] = __builtin_amdgcn_wmma_f32_16x16x32_bf16(false, a, false, b, (short)0, acc[j],
                                                       false, false);
    }
  }
  const int mb = (lane >> 4) << 3;
  const int n  = lane & 15;
  #pragma unroll
  for (int j = 0; j < 4; ++j) {
    int col = n0 + j * 16 + n;
    float bias = bo[col];
    #pragma unroll
    for (int v = 0; v < 8; ++v) {
      int row = mt * 16 + mb + v;
      out[(size_t)row * NC + col] = acc[j][v] + bias;
    }
  }
}

// ---------------- host launch ----------------
extern "C" void kernel_launch(void* const* d_in, const int* in_sizes, int n_in,
                              void* d_out, int out_size, void* d_ws, size_t ws_size,
                              hipStream_t stream) {
  (void)in_sizes; (void)n_in; (void)out_size; (void)ws_size;
  const float* x  = (const float*)d_in[0];
  const float* Wq = (const float*)d_in[1];
  const float* Wk = (const float*)d_in[2];
  const float* Wv = (const float*)d_in[3];
  const float* Wo = (const float*)d_in[4];
  const float* bo = (const float*)d_in[5];
  float* out = (float*)d_out;

  char* ws = (char*)d_ws;
  constexpr size_t SZ_XB  = (size_t)NB * NT * NC * 2;      // 8 MiB  (x as bf16)
  constexpr size_t SZ_WT  = (size_t)NC * NC * 2;           // 128 KiB per weight
  constexpr size_t SZ_QKV = (size_t)NB * NH * NT * ND * 2; // 8 MiB each
  bf16_t* xb  = (bf16_t*)(ws);
  bf16_t* Wqt = (bf16_t*)(ws + SZ_XB);
  bf16_t* Wkt = (bf16_t*)(ws + SZ_XB + SZ_WT);
  bf16_t* Wvt = (bf16_t*)(ws + SZ_XB + 2 * SZ_WT);
  bf16_t* Wot = (bf16_t*)(ws + SZ_XB + 3 * SZ_WT);
  bf16_t* Qb  = (bf16_t*)(ws + SZ_XB + 4 * SZ_WT);
  bf16_t* Kb  = (bf16_t*)(ws + SZ_XB + 4 * SZ_WT + SZ_QKV);
  bf16_t* Vt  = (bf16_t*)(ws + SZ_XB + 4 * SZ_WT + 2 * SZ_QKV);
  bf16_t* att = (bf16_t*)(ws + SZ_XB + 4 * SZ_WT + 3 * SZ_QKV);

  const int nx = NB * NT * NC;
  cvt_f32_bf16<<<(nx + 255) / 256, 256, 0, stream>>>(x, xb, nx);
  wt_transpose<<<256, 256, 0, stream>>>(Wq, Wqt);
  wt_transpose<<<256, 256, 0, stream>>>(Wk, Wkt);
  wt_transpose<<<256, 256, 0, stream>>>(Wv, Wvt);
  wt_transpose<<<256, 256, 0, stream>>>(Wo, Wot);

  qkv_gemm<<<dim3(NB * NT / 16, 12), 32, 0, stream>>>(xb, Wqt, Wkt, Wvt, Qb, Kb, Vt);
  attn_kernel<<<dim3(NB * NH, NT / 64), 128, 0, stream>>>(Qb, Kb, Vt, att);
  out_gemm<<<dim3(NB * NT / 16, 4), 32, 0, stream>>>(att, Wot, bo, out);
}
